// MappingNetwork_22445499089154
// MI455X (gfx1250) — matrix operand order repaired and checked by reading, not compiled, not run on hardware
//
#include <hip/hip_runtime.h>

typedef __attribute__((ext_vector_type(16))) __bf16 v16bf;
typedef __attribute__((ext_vector_type(8)))  float  v8f;

#define B_TOT   8192
#define HID     512
#define LATENT  16
#define STYLE   64
#define NCLS    10
#define TILE_M  64
#define MAXT    138   // 8192/64 + 10 classes worth of tail tiles

#if defined(__has_builtin)
#if __has_builtin(__builtin_amdgcn_sched_barrier)
#define SCHED_FENCE() __builtin_amdgcn_sched_barrier(0)
#endif
#endif
#ifndef SCHED_FENCE
#define SCHED_FENCE() asm volatile("" ::: "memory")
#endif

struct __attribute__((aligned(16))) U128 { unsigned a, b, c, d; };
struct __attribute__((aligned(16))) U256 { U128 lo, hi; };

static __device__ __forceinline__ unsigned short f2bf(float f) {
    // round-to-nearest-even fp32 -> bf16
    unsigned u = __float_as_uint(f);
    unsigned r = u + 0x7FFFu + ((u >> 16) & 1u);
    return (unsigned short)(r >> 16);
}

static __device__ __forceinline__ v16bf load_frag(const unsigned short* p_lo,
                                                  const unsigned short* p_hi) {
    U256 t;
    t.lo = *(const U128*)p_lo;   // 8 bf16, 16B aligned -> ds_load_b128
    t.hi = *(const U128*)p_hi;
    return __builtin_bit_cast(v16bf, t);
}

// ---------------------------------------------------------------------------
// Async global->LDS copies (CDNA5 GLOBAL_LOAD_ASYNC_TO_LDS, ASYNCcnt-tracked).
// INST_OFFSET is added to BOTH the LDS and global addresses per the ISA, so a
// single (vdst, vaddr) pair covers the whole slab with immediate offsets.
// ---------------------------------------------------------------------------
static __device__ __forceinline__ void async_copy32k(const unsigned short* g,
                                                     unsigned short* l, int tid) {
    unsigned loff = (unsigned)(size_t)(void*)l + (unsigned)tid * 16u;
    unsigned goff = (unsigned)tid * 16u;
    asm volatile(
        "global_load_async_to_lds_b128 %0, %1, %2\n\t"
        "global_load_async_to_lds_b128 %0, %1, %2 offset:4096\n\t"
        "global_load_async_to_lds_b128 %0, %1, %2 offset:8192\n\t"
        "global_load_async_to_lds_b128 %0, %1, %2 offset:12288\n\t"
        "global_load_async_to_lds_b128 %0, %1, %2 offset:16384\n\t"
        "global_load_async_to_lds_b128 %0, %1, %2 offset:20480\n\t"
        "global_load_async_to_lds_b128 %0, %1, %2 offset:24576\n\t"
        "global_load_async_to_lds_b128 %0, %1, %2 offset:28672"
        :: "v"(loff), "v"(goff), "s"(g) : "memory");
}

static __device__ __forceinline__ void async_copy4k(const unsigned short* g,
                                                    unsigned short* l, int tid) {
    unsigned loff = (unsigned)(size_t)(void*)l + (unsigned)tid * 16u;
    unsigned goff = (unsigned)tid * 16u;
    asm volatile(
        "global_load_async_to_lds_b128 %0, %1, %2"
        :: "v"(loff), "v"(goff), "s"(g) : "memory");
}

static __device__ __forceinline__ void wait_async(int n) {
    if (n == 0)      asm volatile("s_wait_asynccnt 0x0" ::: "memory");
    else if (n == 1) asm volatile("s_wait_asynccnt 0x1" ::: "memory");
    else             asm volatile("s_wait_asynccnt 0x8" ::: "memory");
}

// ---------------------------------------------------------------------------
// Pre-pass: class histogram + prefix sum + scatter permutation + tile descs.
// ---------------------------------------------------------------------------
__global__ __launch_bounds__(256) void build_order_kernel(
    const int* __restrict__ y, int* __restrict__ order, int* __restrict__ desc) {
    __shared__ int cnt[NCLS], cur[NCLS];
    int t = threadIdx.x;
    if (t < NCLS) cnt[t] = 0;
    __syncthreads();
    for (int b = t; b < B_TOT; b += 256) atomicAdd(&cnt[y[b]], 1);
    __syncthreads();
    if (t == 0) {
        int run = 0, td = 0;
        for (int c = 0; c < NCLS; ++c) {
            cur[c] = run;
            int n = cnt[c];
            for (int i = 0; i < n; i += TILE_M) {
                desc[3 * td + 0] = c;
                desc[3 * td + 1] = run + i;
                int rem = n - i;
                desc[3 * td + 2] = rem < TILE_M ? rem : TILE_M;
                ++td;
            }
            run += n;
        }
        for (; td < MAXT; ++td) {
            desc[3 * td + 0] = 0; desc[3 * td + 1] = 0; desc[3 * td + 2] = 0;
        }
    }
    __syncthreads();
    for (int b = t; b < B_TOT; b += 256) {
        int pos = atomicAdd(&cur[y[b]], 1);   // ds_add_rtn_u32
        order[pos] = b;
    }
}

// ---------------------------------------------------------------------------
// One-time weight conversion: f32 [Kactual x N] (x nmat) -> bf16 in WMMA
// B-slab layout out[mat][s][n][k] with k in 0..31, zero-padded to Kpad rows.
// ---------------------------------------------------------------------------
__global__ __launch_bounds__(256) void convert_slabs_kernel(
    const float* __restrict__ W, unsigned short* __restrict__ out,
    int Kpad, int N, int kvalid, int nmat) {
    size_t total = (size_t)nmat * Kpad * N;
    size_t perMat = (size_t)Kpad * N;
    for (size_t e = blockIdx.x * 256ull + threadIdx.x; e < total;
         e += (size_t)gridDim.x * 256ull) {
        size_t mat = e / perMat;
        size_t r   = e % perMat;
        int s = (int)(r / ((size_t)N * 32));
        int q = (int)(r % ((size_t)N * 32));
        int n = q >> 5, k = q & 31;
        int kk = s * 32 + k;
        float v = 0.f;
        if (kk < kvalid) v = W[mat * ((size_t)kvalid * N) + (size_t)kk * N + n];
        out[e] = f2bf(v);
    }
}

// ===========================================================================
// FAST PATH: weights pre-slabbed bf16 in d_ws; inner loop streams 32KB slabs
// with double-buffered async global->LDS copies overlapped with WMMA.
// 64x512 tile per workgroup; 8 waves = 4 rowgroups x 2 colgroups (256 cols).
// B-fragments are software-pipelined 2 ahead (3-slot rotation) so each WMMA
// has ~2 WMMAs of ds_load slack instead of a serialized LDS round-trip.
// ===========================================================================
static __device__ __forceinline__ void hidden_layer_fast(
    int tid, const unsigned short* src, unsigned short* dst,
    const unsigned short* __restrict__ wbase, const float* __restrict__ bias,
    int nslab, unsigned short* wt0, unsigned short* wt1) {
    int lane = tid & 31, wave = tid >> 5;
    int half = lane >> 4, m = lane & 15;
    int rg = wave >> 1, cg = wave & 1;

    v8f acc[16];
#pragma unroll
    for (int j = 0; j < 16; ++j) acc[j] = {};

    async_copy32k(wbase, wt0, tid);
    for (int s = 0; s < nslab; ++s) {
        unsigned short* curw = (s & 1) ? wt1 : wt0;
        unsigned short* nxtw = (s & 1) ? wt0 : wt1;
        bool more = (s + 1 < nslab);
        if (more) async_copy32k(wbase + (size_t)(s + 1) * (HID * 32), nxtw, tid);
        wait_async(more ? 8 : 0);
        __syncthreads();

        int k0 = s * 32;
        int arow = rg * 16 + m;
        const unsigned short* ap = src + arow * HID + k0 + half * 16;
        v16bf a = load_frag(ap, ap + 8);

        const unsigned short* bbase = curw + (cg * 256 + m) * 32 + half * 16;
        v16bf bb[3];
        bb[0] = load_frag(bbase, bbase + 8);
        bb[1] = load_frag(bbase + 16 * 32, bbase + 16 * 32 + 8);
#pragma unroll
        for (int ct = 0; ct < 16; ++ct) {
            if (ct + 2 < 16) {
                const unsigned short* bp = bbase + (ct + 2) * 16 * 32;
                bb[(ct + 2) % 3] = load_frag(bp, bp + 8);
            }
            acc[ct] = __builtin_amdgcn_wmma_f32_16x16x32_bf16(
                false, a, false, bb[ct % 3], (short)0, acc[ct], false, false);
            if ((ct & 3) == 3) SCHED_FENCE();
        }
        __syncthreads();
    }
#pragma unroll
    for (int ct = 0; ct < 16; ++ct) {
        int col = cg * 256 + ct * 16 + m;
        float bv = bias[col];
#pragma unroll
        for (int v = 0; v < 8; ++v) {
            int row = rg * 16 + half * 8 + v;
            float x = acc[ct][v] + bv;
            x = x > 0.f ? x : 0.f;
            dst[row * HID + col] = f2bf(x);
        }
    }
}

__global__ __launch_bounds__(256) void mapping_main_fast(
    const float* __restrict__ z, const int* __restrict__ order,
    const int* __restrict__ desc, const unsigned short* __restrict__ wbf,
    const float* __restrict__ sb0, const float* __restrict__ sb1,
    const float* __restrict__ sb2, const float* __restrict__ sb3,
    const float* __restrict__ ub0, const float* __restrict__ ub1,
    const float* __restrict__ ub2, const float* __restrict__ ub3,
    float* __restrict__ out) {
    extern __shared__ __align__(16) unsigned short sh[];
    int bid = blockIdx.x, tid = threadIdx.x;
    int cls   = desc[3 * bid + 0];
    int start = desc[3 * bid + 1];
    int cnt   = desc[3 * bid + 2];
    if (cnt == 0) return;

    unsigned short* bufA = sh;                       // 64x512 bf16 = 64KB
    unsigned short* bufB = sh + TILE_M * HID;        // 64KB
    unsigned short* wt0  = sh + 2 * TILE_M * HID;    // 32KB
    unsigned short* wt1  = wt0 + HID * 32;           // 32KB

    // stage z tile into bufA[r][0..31], zero-padded
#pragma unroll
    for (int i = 0; i < 8; ++i) {
        int idx = tid + (i << 8);        // 0..2047
        int r = idx >> 5, c = idx & 31;
        unsigned short v = 0;
        if (r < cnt && c < LATENT) v = f2bf(z[(size_t)order[start + r] * LATENT + c]);
        bufA[r * HID + c] = v;
    }

    // ws offsets (bf16 elements) of the pre-slabbed weights
    const unsigned short* w_sw0 = wbf;                           // 1 slab, padded K
    const unsigned short* w_sw1 = wbf + 16384;
    const unsigned short* w_sw2 = wbf + 278528;
    const unsigned short* w_sw3 = wbf + 540672;
    const unsigned short* w_uw0 = wbf + 802816  + (size_t)cls * 262144;
    const unsigned short* w_uw1 = wbf + 3424256 + (size_t)cls * 262144;
    const unsigned short* w_uw2 = wbf + 6045696 + (size_t)cls * 262144;
    const unsigned short* w_uw3 = wbf + 8667136 + (size_t)cls * 32768;

    hidden_layer_fast(tid, bufA, bufB, w_sw0, sb0, 1,  wt0, wt1);
    hidden_layer_fast(tid, bufB, bufA, w_sw1, sb1, 16, wt0, wt1);
    hidden_layer_fast(tid, bufA, bufB, w_sw2, sb2, 16, wt0, wt1);
    hidden_layer_fast(tid, bufB, bufA, w_sw3, sb3, 16, wt0, wt1);
    hidden_layer_fast(tid, bufA, bufB, w_uw0, ub0 + cls * HID, 16, wt0, wt1);
    hidden_layer_fast(tid, bufB, bufA, w_uw1, ub1 + cls * HID, 16, wt0, wt1);
    hidden_layer_fast(tid, bufA, bufB, w_uw2, ub2 + cls * HID, 16, wt0, wt1);

    // final 512->64: 4KB slabs, double-buffered, no ReLU, scatter f32 rows.
    // 64x64 output = 4 rowgroups x 4 coltiles; 2 tiles per wave.
    {
        const float* bf = ub3 + cls * STYLE;
        int lane = tid & 31, wave = tid >> 5;
        int half = lane >> 4, m = lane & 15;
        int rg = wave >> 1, ch = wave & 1;
        v8f acc[2]; acc[0] = {}; acc[1] = {};
        async_copy4k(w_uw3, wt0, tid);
        for (int s = 0; s < 16; ++s) {
            unsigned short* curw = (s & 1) ? wt1 : wt0;
            unsigned short* nxtw = (s & 1) ? wt0 : wt1;
            bool more = (s + 1 < 16);
            if (more) async_copy4k(w_uw3 + (size_t)(s + 1) * (STYLE * 32), nxtw, tid);
            wait_async(more ? 1 : 0);
            __syncthreads();
            int k0 = s * 32;
            int arow = rg * 16 + m;
            const unsigned short* ap = bufB + arow * HID + k0 + half * 16;
            v16bf a = load_frag(ap, ap + 8);
            const unsigned short* bp0 = curw + (ch * 32 + m) * 32 + half * 16;
            const unsigned short* bp1 = bp0 + 16 * 32;
            v16bf b0 = load_frag(bp0, bp0 + 8);
            v16bf b1 = load_frag(bp1, bp1 + 8);
            acc[0] = __builtin_amdgcn_wmma_f32_16x16x32_bf16(
                false, a, false, b0, (short)0, acc[0], false, false);
            acc[1] = __builtin_amdgcn_wmma_f32_16x16x32_bf16(
                false, a, false, b1, (short)0, acc[1], false, false);
            __syncthreads();
        }
#pragma unroll
        for (int ct = 0; ct < 2; ++ct) {
            int col = ch * 32 + ct * 16 + m;
            float bv = bf[col];
#pragma unroll
            for (int v = 0; v < 8; ++v) {
                int r = rg * 16 + half * 8 + v;
                if (r < cnt) out[(size_t)order[start + r] * STYLE + col] = acc[ct][v] + bv;
            }
        }
    }
}

// ===========================================================================
// FALLBACK PATH (small d_ws): convert f32 weight slabs in the main loop.
// ===========================================================================
static __device__ __forceinline__ void load_slab512(int tid, const float* __restrict__ Wg,
                                                    int k0, int kvalid,
                                                    unsigned short* wt) {
#pragma unroll 4
    for (int i = 0; i < 32; ++i) {
        int p = tid + (i << 8);
        int k = p >> 8;
        int n = (p & 255) << 1;
        float x = 0.f, x1 = 0.f;
        int kk = k0 + k;
        if (kk < kvalid) {
            const float* s = Wg + (size_t)kk * HID + n;
            x = s[0]; x1 = s[1];
        }
        wt[n * 32 + k]       = f2bf(x);
        wt[(n + 1) * 32 + k] = f2bf(x1);
    }
}

static __device__ __forceinline__ void load_slab64(int tid, const float* __restrict__ Wg,
                                                   int k0, unsigned short* wt) {
#pragma unroll
    for (int i = 0; i < 4; ++i) {
        int p = tid + (i << 8);
        int k = p >> 5;
        int n = (p & 31) << 1;
        const float* s = Wg + (size_t)(k0 + k) * STYLE + n;
        wt[n * 32 + k]       = f2bf(s[0]);
        wt[(n + 1) * 32 + k] = f2bf(s[1]);
    }
}

static __device__ __forceinline__ void hidden_layer(
    int tid, const unsigned short* src, unsigned short* dst,
    const float* __restrict__ Wg, const float* __restrict__ bias,
    int K, int kvalid, unsigned short* wt) {
    int lane = tid & 31, wave = tid >> 5;
    int half = lane >> 4, m = lane & 15;
    int rg = wave >> 1, cg = wave & 1;

    v8f acc[16];
#pragma unroll
    for (int j = 0; j < 16; ++j) acc[j] = {};

    for (int k0 = 0; k0 < K; k0 += 32) {
        load_slab512(tid, Wg, k0, kvalid, wt);
        __syncthreads();
        int arow = rg * 16 + m;
        const unsigned short* ap = src + arow * HID + k0 + half * 16;
        v16bf a = load_frag(ap, ap + 8);

        const unsigned short* bbase = wt + (cg * 256 + m) * 32 + half * 16;
        v16bf bb[3];
        bb[0] = load_frag(bbase, bbase + 8);
        bb[1] = load_frag(bbase + 16 * 32, bbase + 16 * 32 + 8);
#pragma unroll
        for (int ct = 0; ct < 16; ++ct) {
            if (ct + 2 < 16) {
                const unsigned short* bp = bbase + (ct + 2) * 16 * 32;
                bb[(ct + 2) % 3] = load_frag(bp, bp + 8);
            }
            acc[ct] = __builtin_amdgcn_wmma_f32_16x16x32_bf16(
                false, a, false, bb[ct % 3], (short)0, acc[ct], false, false);
            if ((ct & 3) == 3) SCHED_FENCE();
        }
        __syncthreads();
    }
#pragma unroll
    for (int ct = 0; ct < 16; ++ct) {
        int col = cg * 256 + ct * 16 + m;
        float bv = bias[col];
#pragma unroll
        for (int v = 0; v < 8; ++v) {
            int row = rg * 16 + half * 8 + v;
            float x = acc[ct][v] + bv;
            x = x > 0.f ? x : 0.f;
            dst[row * HID + col] = f2bf(x);
        }
    }
}

__global__ __launch_bounds__(256) void mapping_main(
    const float* __restrict__ z, const int* __restrict__ order,
    const int* __restrict__ desc,
    const float* __restrict__ sw0, const float* __restrict__ sb0,
    const float* __restrict__ sw1, const float* __restrict__ sb1,
    const float* __restrict__ sw2, const float* __restrict__ sb2,
    const float* __restrict__ sw3, const float* __restrict__ sb3,
    const float* __restrict__ uw0, const float* __restrict__ ub0,
    const float* __restrict__ uw1, const float* __restrict__ ub1,
    const float* __restrict__ uw2, const float* __restrict__ ub2,
    const float* __restrict__ uw3, const float* __restrict__ ub3,
    float* __restrict__ out) {
    extern __shared__ __align__(16) unsigned short sh[];
    int bid = blockIdx.x, tid = threadIdx.x;
    int cls   = desc[3 * bid + 0];
    int start = desc[3 * bid + 1];
    int cnt   = desc[3 * bid + 2];
    if (cnt == 0) return;

    unsigned short* bufA = sh;
    unsigned short* bufB = sh + TILE_M * HID;
    unsigned short* wt   = sh + 2 * TILE_M * HID;

#pragma unroll
    for (int i = 0; i < 8; ++i) {
        int idx = tid + (i << 8);
        int r = idx >> 5, c = idx & 31;
        unsigned short v = 0;
        if (r < cnt && c < LATENT) v = f2bf(z[(size_t)order[start + r] * LATENT + c]);
        bufA[r * HID + c] = v;
    }

    hidden_layer(tid, bufA, bufB, sw0, sb0, 32, LATENT, wt);
    hidden_layer(tid, bufB, bufA, sw1, sb1, HID, HID, wt);
    hidden_layer(tid, bufA, bufB, sw2, sb2, HID, HID, wt);
    hidden_layer(tid, bufB, bufA, sw3, sb3, HID, HID, wt);
    hidden_layer(tid, bufA, bufB, uw0 + (size_t)cls * HID * HID, ub0 + cls * HID, HID, HID, wt);
    hidden_layer(tid, bufB, bufA, uw1 + (size_t)cls * HID * HID, ub1 + cls * HID, HID, HID, wt);
    hidden_layer(tid, bufA, bufB, uw2 + (size_t)cls * HID * HID, ub2 + cls * HID, HID, HID, wt);

    {
        const float* Wf = uw3 + (size_t)cls * HID * STYLE;
        const float* bf = ub3 + cls * STYLE;
        int lane = tid & 31, wave = tid >> 5;
        int half = lane >> 4, m = lane & 15;
        int rg = wave >> 1, ch = wave & 1;
        v8f acc[2]; acc[0] = {}; acc[1] = {};
        for (int k0 = 0; k0 < HID; k0 += 32) {
            load_slab64(tid, Wf, k0, wt);
            __syncthreads();
            int arow = rg * 16 + m;
            const unsigned short* ap = bufB + arow * HID + k0 + half * 16;
            v16bf a = load_frag(ap, ap + 8);
            const unsigned short* bp0 = wt + (ch * 32 + m) * 32 + half * 16;
            const unsigned short* bp1 = bp0 + 16 * 32;
            v16bf b0 = load_frag(bp0, bp0 + 8);
            v16bf b1 = load_frag(bp1, bp1 + 8);
            acc[0] = __builtin_amdgcn_wmma_f32_16x16x32_bf16(
                false, a, false, b0, (short)0, acc[0], false, false);
            acc[1] = __builtin_amdgcn_wmma_f32_16x16x32_bf16(
                false, a, false, b1, (short)0, acc[1], false, false);
            __syncthreads();
        }
#pragma unroll
        for (int ct = 0; ct < 2; ++ct) {
            int col = ch * 32 + ct * 16 + m;
            float bv = bf[col];
#pragma unroll
            for (int v = 0; v < 8; ++v) {
                int r = rg * 16 + half * 8 + v;
                if (r < cnt) out[(size_t)order[start + r] * STYLE + col] = acc[ct][v] + bv;
            }
        }
    }
}

// ---------------------------------------------------------------------------
extern "C" void kernel_launch(void* const* d_in, const int* in_sizes, int n_in,
                              void* d_out, int out_size, void* d_ws, size_t ws_size,
                              hipStream_t stream) {
    (void)in_sizes; (void)n_in; (void)out_size;
    const float* z   = (const float*)d_in[0];
    const int*   y   = (const int*)d_in[1];
    const float* sw0 = (const float*)d_in[2];  const float* sb0 = (const float*)d_in[3];
    const float* sw1 = (const float*)d_in[4];  const float* sb1 = (const float*)d_in[5];
    const float* sw2 = (const float*)d_in[6];  const float* sb2 = (const float*)d_in[7];
    const float* sw3 = (const float*)d_in[8];  const float* sb3 = (const float*)d_in[9];
    const float* uw0 = (const float*)d_in[10]; const float* ub0 = (const float*)d_in[11];
    const float* uw1 = (const float*)d_in[12]; const float* ub1 = (const float*)d_in[13];
    const float* uw2 = (const float*)d_in[14]; const float* ub2 = (const float*)d_in[15];
    const float* uw3 = (const float*)d_in[16]; const float* ub3 = (const float*)d_in[17];

    int* order = (int*)d_ws;          // 8192 ints
    int* desc  = order + B_TOT;       // MAXT*3 ints

    build_order_kernel<<<1, 256, 0, stream>>>(y, order, desc);

    const size_t intBytes = 36864;                 // order+desc, 16B-aligned pad
    const size_t wElems   = 8994816ull;            // total pre-slabbed bf16 elems
    if (ws_size >= intBytes + wElems * 2) {
        unsigned short* wbf = (unsigned short*)((char*)d_ws + intBytes);
        // one-time f32 -> bf16 slab-layout conversion (weights then live in L2)
        convert_slabs_kernel<<<64,   256, 0, stream>>>(sw0, wbf + 0,       32,  HID,   LATENT, 1);
        convert_slabs_kernel<<<1024, 256, 0, stream>>>(sw1, wbf + 16384,   HID, HID,   HID,    1);
        convert_slabs_kernel<<<1024, 256, 0, stream>>>(sw2, wbf + 278528,  HID, HID,   HID,    1);
        convert_slabs_kernel<<<1024, 256, 0, stream>>>(sw3, wbf + 540672,  HID, HID,   HID,    1);
        convert_slabs_kernel<<<2048, 256, 0, stream>>>(uw0, wbf + 802816,  HID, HID,   HID,    NCLS);
        convert_slabs_kernel<<<2048, 256, 0, stream>>>(uw1, wbf + 3424256, HID, HID,   HID,    NCLS);
        convert_slabs_kernel<<<2048, 256, 0, stream>>>(uw2, wbf + 6045696, HID, HID,   HID,    NCLS);
        convert_slabs_kernel<<<512,  256, 0, stream>>>(uw3, wbf + 8667136, HID, STYLE, HID,    NCLS);

        size_t lds_bytes = (size_t)(2 * TILE_M * HID + 2 * HID * 32) * sizeof(unsigned short); // 192KB
        mapping_main_fast<<<MAXT, 256, lds_bytes, stream>>>(
            z, order, desc, wbf, sb0, sb1, sb2, sb3, ub0, ub1, ub2, ub3, (float*)d_out);
    } else {
        size_t lds_bytes = (size_t)(2 * TILE_M * HID + HID * 32) * sizeof(unsigned short); // 160KB
        mapping_main<<<MAXT, 256, lds_bytes, stream>>>(
            z, order, desc, sw0, sb0, sw1, sb1, sw2, sb2, sw3, sb3,
            uw0, ub0, uw1, ub1, uw2, ub2, uw3, ub3, (float*)d_out);
    }
}
